// CrossEntropyLoss_56942676410880
// MI455X (gfx1250) — matrix operand order repaired and checked by reading, not compiled, not run on hardware
//
#include <hip/hip_runtime.h>
#include <hip/hip_bf16.h>
#include <stdint.h>

// ---------------- problem constants ----------------
#define BATCH     4
#define SEQ       2048
#define DMODEL    2048
#define VOCAB     32000
#define NROWS     (BATCH * (SEQ - 1))   // 8188 shifted token rows
#define ROWPAD    8192                  // padded row count (zero-filled tail)
#define IGNORE_IX (-100)

#define MTILE     64                    // rows per workgroup (A tile in LDS)
#define ROWBLKS   (ROWPAD / MTILE)      // 128
#define NSPLIT    4                     // column splits per row block
#define NTILES    (VOCAB / 16)          // 2000 column tiles of 16
#define NPAIRS    (NTILES / 2)          // 1000 pairs of column tiles
#define PAIRS_PER_SPLIT (NPAIRS / NSPLIT) // 250

typedef __attribute__((ext_vector_type(16))) __bf16   v16bf;
typedef __attribute__((ext_vector_type(8)))  float    v8f;
typedef __attribute__((ext_vector_type(8)))  unsigned int u32x8;

// ---------------- helpers ----------------
__device__ __forceinline__ unsigned int pack2bf16(float x0, float x1) {
  unsigned int u0 = __float_as_uint(x0);
  unsigned int u1 = __float_as_uint(x1);
  u0 = (u0 + 0x7FFFu + ((u0 >> 16) & 1u)) >> 16;   // RNE to bf16
  u1 = (u1 + 0x7FFFu + ((u1 >> 16) & 1u)) >> 16;
  return u0 | (u1 << 16);
}

__device__ __forceinline__ float bf_lo(unsigned int u) { return __uint_as_float(u << 16); }
__device__ __forceinline__ float bf_hi(unsigned int u) { return __uint_as_float(u & 0xFFFF0000u); }

__device__ __forceinline__ void lse_merge(float& mx, float& s, float omx, float os) {
  float nm = fmaxf(mx, omx);
  s = s * __expf(mx - nm) + os * __expf(omx - nm);
  mx = nm;
}

// ---------------- kernel 0: zero accumulators ----------------
__global__ void ce_init_kernel(float* acc) {
  acc[0] = 0.0f;   // sum of nll
  acc[1] = 0.0f;   // valid count
}

// ---------------- kernel 1: weight fp32 -> bf16 ----------------
__global__ void ce_convw_kernel(const float* __restrict__ W, unsigned int* __restrict__ Wb) {
  const long long npairs = (long long)VOCAB * DMODEL / 2;   // 32,768,000
  long long stride = (long long)gridDim.x * blockDim.x;
  for (long long p = (long long)blockIdx.x * blockDim.x + threadIdx.x; p < npairs; p += stride) {
    float2 f = ((const float2*)W)[p];
    Wb[p] = pack2bf16(f.x, f.y);
  }
}

// ---------------- kernel 2: shifted hidden fp32 -> bf16 (row-packed, zero-padded) ----------------
__global__ void ce_convh_kernel(const float* __restrict__ H, unsigned int* __restrict__ Hb) {
  const long long npairs = (long long)ROWPAD * (DMODEL / 2);  // 8,388,608
  long long stride = (long long)gridDim.x * blockDim.x;
  for (long long p = (long long)blockIdx.x * blockDim.x + threadIdx.x; p < npairs; p += stride) {
    int n  = (int)(p >> 10);          // row (1024 pairs per row)
    int pc = (int)(p & 1023);
    unsigned int out = 0u;
    if (n < NROWS) {
      int b = n / (SEQ - 1);
      int s = n - b * (SEQ - 1);
      const float* src = H + ((size_t)b * SEQ + s) * DMODEL + (size_t)pc * 2;
      out = pack2bf16(src[0], src[1]);
    }
    Hb[p] = out;
  }
}

// ---------------- kernel 3: fused WMMA GEMM + online logsumexp ----------------
// grid = (ROWBLKS, NSPLIT), block = 256 (8 waves of 32)
// Each wave iteration covers a 64-row x 32-col tile: 4 M-subtiles x 2 N-tiles,
// so each A fragment (ds_load_b128 x2) feeds TWO wmma ops -> 1 ds_b128 per wmma.
// No label handling here: the picked-label logit is recomputed in the merge pass,
// keeping the hot loop to loads + wmma + online max/sum only (lower VGPR pressure).
__global__ __launch_bounds__(256)
void ce_gemm_lse_kernel(const uint16_t* __restrict__ Hb,
                        const uint16_t* __restrict__ Wb,
                        float* __restrict__ part) {
  extern __shared__ char smem_raw[];
  // LDS layout: A tile 64x2048 bf16 (256KB) | per-wave reductions
  uint4* A4    = (uint4*)smem_raw;                       // 16384 uint4
  float* redMx = (float*)(smem_raw + MTILE * DMODEL * 2);
  float* redS  = redMx + 8 * MTILE;

  const int rowBlock = blockIdx.x;
  const int split    = blockIdx.y;
  const int rowBase  = rowBlock * MTILE;
  const int lane     = threadIdx.x & 31;
  const int w        = threadIdx.x >> 5;
  const int half     = lane >> 4;       // 0: lanes 0-15, 1: lanes 16-31
  const int nIdx     = lane & 15;       // column-within-tile index (B/C layout)
  const int mIdx     = lane & 15;       // row-within-subtile index (A layout)

  // stage 64 rows of bf16 H into LDS (contiguous)
  {
    const uint4* src4 = (const uint4*)(Hb + (size_t)rowBase * DMODEL);
    for (int i = threadIdx.x; i < MTILE * DMODEL / 8; i += 256) A4[i] = src4[i];
  }
  __syncthreads();

  // online-softmax state per (subtile, C-vgpr) slot (row-indexed; columns stream through)
  float mx[4][8], sm[4][8];
#pragma unroll
  for (int st = 0; st < 4; ++st)
#pragma unroll
    for (int r = 0; r < 8; ++r) { mx[st][r] = -INFINITY; sm[st][r] = 0.0f; }

  const uint4* Wb4 = (const uint4*)Wb;

  // ---- column tile-pair loop (this wave's share of this split) ----
  for (int tp = split * PAIRS_PER_SPLIT + w; tp < (split + 1) * PAIRS_PER_SPLIT; tp += 8) {
    const int col0 = tp * 32;

    v8f c[2][4];
#pragma unroll
    for (int nt = 0; nt < 2; ++nt)
#pragma unroll
      for (int st = 0; st < 4; ++st)
#pragma unroll
        for (int r = 0; r < 8; ++r) c[nt][st][r] = 0.0f;

    // B fragment bases (uint4 units): W rows col0+nIdx and col0+16+nIdx, k start = half*16
    const size_t bRow0 = (size_t)(col0 + nIdx) * (DMODEL / 8) + (size_t)half * 2;
    const size_t bRow1 = bRow0 + (size_t)16 * (DMODEL / 8);

#pragma unroll 2
    for (int k0 = 0; k0 < DMODEL; k0 += 32) {
      // B: 32x16 bf16 per tile, lane holds one W row, k = k0 + half*16 .. +15
      size_t bi0 = bRow0 + (size_t)(k0 >> 3);
      size_t bi1 = bRow1 + (size_t)(k0 >> 3);
      uint4 b0a = Wb4[bi0], b0b = Wb4[bi0 + 1];
      uint4 b1a = Wb4[bi1], b1b = Wb4[bi1 + 1];
      u32x8 bb0, bb1;
      bb0[0] = b0a.x; bb0[1] = b0a.y; bb0[2] = b0a.z; bb0[3] = b0a.w;
      bb0[4] = b0b.x; bb0[5] = b0b.y; bb0[6] = b0b.z; bb0[7] = b0b.w;
      bb1[0] = b1a.x; bb1[1] = b1a.y; bb1[2] = b1a.z; bb1[3] = b1a.w;
      bb1[4] = b1b.x; bb1[5] = b1b.y; bb1[6] = b1b.z; bb1[7] = b1b.w;
      v16bf bf0 = __builtin_bit_cast(v16bf, bb0);
      v16bf bf1 = __builtin_bit_cast(v16bf, bb1);

#pragma unroll
      for (int st = 0; st < 4; ++st) {
        // A: 16x32 bf16, lane holds row st*16+mIdx, k = k0 + half*16 .. +15
        int ai = (st * 16 + mIdx) * (DMODEL / 8) + (k0 >> 3) + half * 2;
        uint4 a0 = A4[ai];
        uint4 a1 = A4[ai + 1];   // +16 bytes -> k offset +8
        u32x8 aa;
        aa[0] = a0.x; aa[1] = a0.y; aa[2] = a0.z; aa[3] = a0.w;
        aa[4] = a1.x; aa[5] = a1.y; aa[6] = a1.z; aa[7] = a1.w;
        v16bf af = __builtin_bit_cast(v16bf, aa);

        c[0][st] = __builtin_amdgcn_wmma_f32_16x16x32_bf16(
            false, af, false, bf0, (short)0, c[0][st], false, false);
        c[1][st] = __builtin_amdgcn_wmma_f32_16x16x32_bf16(
            false, af, false, bf1, (short)0, c[1][st], false, false);
      }
    }

    // online logsumexp update: slot (st,r) = row st*16+half*8+r, cols col0+nt*16+nIdx
#pragma unroll
    for (int nt = 0; nt < 2; ++nt)
#pragma unroll
      for (int st = 0; st < 4; ++st)
#pragma unroll
        for (int r = 0; r < 8; ++r) {
          float x  = c[nt][st][r];
          float nm = fmaxf(mx[st][r], x);
          sm[st][r] = sm[st][r] * __expf(mx[st][r] - nm) + __expf(x - nm);
          mx[st][r] = nm;
        }
  }

  // ---- butterfly reduce across the 16 lanes sharing each row (masks stay in-half) ----
#pragma unroll
  for (int st = 0; st < 4; ++st)
#pragma unroll
    for (int r = 0; r < 8; ++r) {
#pragma unroll
      for (int mask = 1; mask <= 8; mask <<= 1) {
        float omx = __shfl_xor(mx[st][r], mask, 32);
        float os  = __shfl_xor(sm[st][r], mask, 32);
        lse_merge(mx[st][r], sm[st][r], omx, os);
      }
    }

  if ((lane & 15) == 0) {
#pragma unroll
    for (int st = 0; st < 4; ++st)
#pragma unroll
      for (int r = 0; r < 8; ++r) {
        int m = st * 16 + half * 8 + r;
        redMx[w * MTILE + m] = mx[st][r];
        redS [w * MTILE + m] = sm[st][r];
      }
  }
  __syncthreads();

  // ---- combine 8 wave partials, emit per-split partial (mx, sum) ----
  if (threadIdx.x < MTILE) {
    int m = threadIdx.x;
    float M = redMx[m], S = redS[m];
#pragma unroll
    for (int w2 = 1; w2 < 8; ++w2)
      lse_merge(M, S, redMx[w2 * MTILE + m], redS[w2 * MTILE + m]);
    size_t idx = ((size_t)(rowBlock * NSPLIT + split) * MTILE + m) * 2;
    part[idx + 0] = M;
    part[idx + 1] = S;
  }
}

// ---------------- kernel 4: merge splits + recompute label logit + accumulate ----------------
// One wave per token row: coalesced packed-bf16 dot(h_row, W[label]) + shfl reduce.
__global__ __launch_bounds__(256)
void ce_merge_kernel(const float* __restrict__ part,
                     const unsigned int* __restrict__ HbU,
                     const unsigned int* __restrict__ WbU,
                     const int* __restrict__ labels,
                     float* __restrict__ acc) {
  const int wv   = (int)((blockIdx.x * blockDim.x + threadIdx.x) >> 5);  // wave id = row
  const int lane = threadIdx.x & 31;
  if (wv >= NROWS) return;                      // uniform within wave
  const int n = wv;
  const int b = n / (SEQ - 1);
  const int s = n - b * (SEQ - 1);
  const int lab = labels[(size_t)b * SEQ + s + 1];
  if (lab == IGNORE_IX) return;                 // uniform within wave

  // picked-label logit: dot over 1024 packed-bf16 dwords, 32 lanes coalesced
  const unsigned int* hrow = HbU + (size_t)n * (DMODEL / 2);
  const unsigned int* wrow = WbU + (size_t)lab * (DMODEL / 2);
  float dot = 0.0f;
#pragma unroll 4
  for (int i = lane; i < DMODEL / 2; i += 32) {
    unsigned int hu = hrow[i];
    unsigned int wu = wrow[i];
    dot = fmaf(bf_lo(hu), bf_lo(wu), dot);
    dot = fmaf(bf_hi(hu), bf_hi(wu), dot);
  }
#pragma unroll
  for (int mask = 16; mask >= 1; mask >>= 1)
    dot += __shfl_xor(dot, mask, 32);

  if (lane == 0) {
    const int rowBlock = n >> 6;
    const int m        = n & 63;
    float M = -INFINITY, S = 0.0f;
#pragma unroll
    for (int sp = 0; sp < NSPLIT; ++sp) {
      size_t idx = ((size_t)(rowBlock * NSPLIT + sp) * MTILE + m) * 2;
      lse_merge(M, S, part[idx + 0], part[idx + 1]);
    }
    float nll = (M + __logf(S)) - dot;
    atomicAdd(&acc[0], nll);
    atomicAdd(&acc[1], 1.0f);
  }
}

// ---------------- kernel 5: finalize mean ----------------
__global__ void ce_final_kernel(const float* acc, float* out) {
  out[0] = acc[0] / fmaxf(acc[1], 1.0f);
}

// ---------------- launcher ----------------
extern "C" void kernel_launch(void* const* d_in, const int* in_sizes, int n_in,
                              void* d_out, int out_size, void* d_ws, size_t ws_size,
                              hipStream_t stream) {
  (void)in_sizes; (void)n_in; (void)out_size; (void)ws_size;
  const float* hidden = (const float*)d_in[0];
  const float* weight = (const float*)d_in[1];
  const int*   labels = (const int*)d_in[2];
  float*       out    = (float*)d_out;

  // workspace layout
  const size_t WBYTES = (size_t)VOCAB * DMODEL * 2;   // 131,072,000 (bf16 W)
  const size_t HBYTES = (size_t)ROWPAD * DMODEL * 2;  //  33,554,432 (bf16 H, padded)
  char* ws   = (char*)d_ws;
  float* acc = (float*)ws;                               // 2 floats @ offset 0
  unsigned int* WbU = (unsigned int*)(ws + 256);
  unsigned int* HbU = (unsigned int*)(ws + 256 + WBYTES);
  float* part       = (float*)(ws + 256 + WBYTES + HBYTES); // 128*4*64*2 floats

  ce_init_kernel<<<1, 1, 0, stream>>>(acc);
  ce_convw_kernel<<<2048, 256, 0, stream>>>(weight, WbU);
  ce_convh_kernel<<<2048, 256, 0, stream>>>(hidden, HbU);

  const size_t smemBytes = (size_t)MTILE * DMODEL * 2       // A tile: 256 KB
                         + 2 * 8 * MTILE * sizeof(float);   // wave reductions
  ce_gemm_lse_kernel<<<dim3(ROWBLKS, NSPLIT), 256, smemBytes, stream>>>(
      (const uint16_t*)HbU, (const uint16_t*)WbU, part);

  ce_merge_kernel<<<(NROWS * 32 + 255) / 256, 256, 0, stream>>>(part, HbU, WbU, labels, acc);
  ce_final_kernel<<<1, 1, 0, stream>>>(acc, out);
}
